// LePEAttention_63075889709089
// MI455X (gfx1250) — matrix operand order repaired and checked by reading, not compiled, not run on hardware
//
#include <hip/hip_runtime.h>

typedef __attribute__((ext_vector_type(32))) __bf16 v32bf;
typedef __attribute__((ext_vector_type(16))) __bf16 v16bf;
typedef __attribute__((ext_vector_type(8)))  __bf16 v8bf;
typedef __attribute__((ext_vector_type(8)))  float  v8f;

#define Bn 16
#define Hh 56
#define Ww 56
#define Cc 256
#define NHEADS 8
#define HD 32
#define WSP 7
#define Ltok 392              // 56*7 tokens per window
#define Ntok 3136             // H*W
#define NC (Ntok*Cc)          // 802816
#define MATSTRIDE ((size_t)Bn*NC)
#define MT 25                 // ceil(392/16) row tiles
#define KPAIRS 13             // ceil(392/32) -> 416 padded keys
#define LPAD 416
#define SCALE_Q 0.17677669529663687f

#define CONCAT16(a, b) __builtin_shufflevector(a, b, 0,1,2,3,4,5,6,7,8,9,10,11,12,13,14,15)

__device__ __forceinline__ v16bf cvt2x8(v8f lo, v8f hi) {
  v8bf a = __builtin_convertvector(lo, v8bf);
  v8bf b = __builtin_convertvector(hi, v8bf);
  return CONCAT16(a, b);
}

// ---------------------------------------------------------------------------
// Kernel 1: LePE = dense 3x3 SAME conv per window, implicit GEMM via WMMA.
// grid (2, 25, 128), block 256 (8 waves); wave -> one 16x16 output tile.
// Branch-free A path: clamped address + 0/1 mask, wide b128 loads.
// ---------------------------------------------------------------------------
__global__ __launch_bounds__(256)
void lepe_conv_kernel(const float* __restrict__ qkv,
                      const float* __restrict__ wconv,
                      const float* __restrict__ bconv,
                      float* __restrict__ out) {
  const int wave = threadIdx.x >> 5;
  const int lane = threadIdx.x & 31;
  const int half = lane >> 4;
  const int l15  = lane & 15;

  const int n0 = (blockIdx.x * 8 + wave) * 16;   // out-channel tile
  const int m0 = blockIdx.y * 16;                // position tile
  const int bw = blockIdx.z;
  const int b  = bw >> 3;
  const int ws = bw & 7;

  const int m  = m0 + l15;                       // this lane's A row (position)
  const int py = m / WSP;
  const int px = m % WSP;
  const bool mvalid = (m < Ltok);

  const float* vbase = qkv + 2*MATSTRIDE + (size_t)b * NC;

  v8f acc = {};
  for (int s = 0; s < 9; ++s) {
    const int ky = s / 3 - 1, kx = s % 3 - 1;
    const int yy = py + ky, xx = px + kx;
    const bool inb = mvalid && (yy >= 0) && (yy < Hh) && (xx >= 0) && (xx < WSP);
    const int yy_c = yy < 0 ? 0 : (yy > Hh - 1 ? Hh - 1 : yy);
    const int xx_c = xx < 0 ? 0 : (xx > WSP - 1 ? WSP - 1 : xx);
    const float msk = inb ? 1.f : 0.f;
    const float* src = vbase + (size_t)(yy_c * Ww + ws * WSP + xx_c) * Cc;
    const float* wsl = wconv + (size_t)s * Cc * Cc;
    for (int c0 = 0; c0 < Cc; c0 += 32) {
      // A: two contiguous 8-float chunks per lane (WMMA-A 16-bit layout)
      v8f alo = *(const v8f*)(src + c0 + 8*half);
      v8f ahi = *(const v8f*)(src + c0 + 16 + 8*half);
      v16bf a = cvt2x8(alo * msk, ahi * msk);

      // B: lane = out-channel column n0+l15, elements over input channels (table k order)
      v16bf bm;
      #pragma unroll
      for (int s2 = 0; s2 < 16; ++s2) {
        const int kk = c0 + (s2 < 8 ? 8*half + s2 : 8 + 8*half + s2);
        bm[s2] = (__bf16)wsl[(size_t)kk * Cc + n0 + l15];
      }

      acc = __builtin_amdgcn_wmma_f32_16x16x32_bf16(false, a, false, bm,
                                                    (short)0, acc, false, false);
    }
  }

  const float bias = bconv[n0 + l15];
  #pragma unroll
  for (int r = 0; r < 8; ++r) {
    const int mrow = m0 + r + 8*half;
    if (mrow < Ltok) {
      const int t = (mrow / WSP) * Ww + ws * WSP + (mrow % WSP);
      out[(size_t)b * NC + (size_t)t * Cc + n0 + l15] = acc[r] + bias;
    }
  }
}

// ---------------------------------------------------------------------------
// Kernel 2: flash attention per (window, head). grid (8, 128), block 256.
// K^T and V staged in LDS as bf16; per-wave 16-row Q tiles; online softmax.
// Accumulates (+=) into out on top of the conv result.
// ---------------------------------------------------------------------------
__global__ __launch_bounds__(256)
void cswin_attn_kernel(const float* __restrict__ qkv,
                       float* __restrict__ out) {
  __shared__ __align__(64) __bf16 KT[HD][LPAD];     // K^T [d][tok]
  __shared__ __align__(64) __bf16 Vs[LPAD][HD];     // V   [tok][d]
  __shared__ __align__(64) __bf16 Ps[8][16][32];    // per-wave P tile

  const int head = blockIdx.x;
  const int bw   = blockIdx.y;
  const int b    = bw >> 3;
  const int ws   = bw & 7;

  const int tid  = threadIdx.x;
  const int wave = tid >> 5;
  const int lane = tid & 31;
  const int half = lane >> 4;
  const int l15  = lane & 15;

  const float* qbase = qkv +                 (size_t)b * NC;
  const float* kbase = qkv +   MATSTRIDE +   (size_t)b * NC;
  const float* vbase = qkv + 2*MATSTRIDE +   (size_t)b * NC;

  // cooperative staging of K^T and V (zero-padded to 416 tokens)
  for (int tok = tid; tok < LPAD; tok += 256) {
    if (tok < Ltok) {
      const int t = (tok / WSP) * Ww + ws * WSP + (tok % WSP);
      const float* kptr = kbase + (size_t)t * Cc + head * HD;
      const float* vptr = vbase + (size_t)t * Cc + head * HD;
      #pragma unroll
      for (int d = 0; d < HD; d += 8) {
        v8f kv = *(const v8f*)(kptr + d);
        v8bf kb = __builtin_convertvector(kv, v8bf);
        #pragma unroll
        for (int i = 0; i < 8; ++i) KT[d + i][tok] = kb[i];   // transpose scatter
        v8f vv = *(const v8f*)(vptr + d);
        *(v8bf*)&Vs[tok][d] = __builtin_convertvector(vv, v8bf);
      }
    } else {
      #pragma unroll
      for (int d = 0; d < HD; ++d) KT[d][tok] = (__bf16)0.f;
      v8bf z = {};
      #pragma unroll
      for (int d = 0; d < HD; d += 8) *(v8bf*)&Vs[tok][d] = z;
    }
  }
  __syncthreads();

  for (int mtile = wave; mtile < MT; mtile += 8) {
    // Q tile in WMMA-A layout, pre-scaled by 1/sqrt(d); branch-free pad
    const int m  = mtile * 16 + l15;
    const int mc = m < Ltok ? m : Ltok - 1;
    const float qmsk = (m < Ltok) ? SCALE_Q : 0.f;
    const int tq = (mc / WSP) * Ww + ws * WSP + (mc % WSP);
    const float* qp = qbase + (size_t)tq * Cc + head * HD;
    v8f qlo = *(const v8f*)(qp + 8*half);
    v8f qhi = *(const v8f*)(qp + 16 + 8*half);
    const v16bf aq = cvt2x8(qlo * qmsk, qhi * qmsk);

    v8f o0 = {}, o1 = {};
    float mx[8], sm[8];
    #pragma unroll
    for (int r = 0; r < 8; ++r) { mx[r] = -1e30f; sm[r] = 0.f; }

    for (int kt = 0; kt < KPAIRS; ++kt) {
      const int n0 = kt * 32;

      // B for S: lane = key token column, elements over head-dim (table k order)
      v16bf b0, b1;
      #pragma unroll
      for (int s2 = 0; s2 < 16; ++s2) {
        const int kk = s2 < 8 ? 8*half + s2 : 8 + 8*half + s2;
        b0[s2] = KT[kk][n0 + l15];
        b1[s2] = KT[kk][n0 + 16 + l15];
      }

      const v8f zz = {};
      v8f s0 = __builtin_amdgcn_wmma_f32_16x16x32_bf16(false, aq, false, b0,
                                                       (short)0, zz, false, false);
      v8f s1 = __builtin_amdgcn_wmma_f32_16x16x32_bf16(false, aq, false, b1,
                                                       (short)0, zz, false, false);

      // mask padded key columns (this lane's columns: n0+l15, n0+16+l15)
      if (n0 + l15 >= Ltok) {
        #pragma unroll
        for (int r = 0; r < 8; ++r) s0[r] = -1e30f;
      }
      if (n0 + 16 + l15 >= Ltok) {
        #pragma unroll
        for (int r = 0; r < 8; ++r) s1[r] = -1e30f;
      }

      // online softmax: rows live at (r, half); columns across 16 lanes
      #pragma unroll
      for (int r = 0; r < 8; ++r) {
        float nm = fmaxf(s0[r], s1[r]);
        nm = fmaxf(nm, __shfl_xor(nm, 1, 32));
        nm = fmaxf(nm, __shfl_xor(nm, 2, 32));
        nm = fmaxf(nm, __shfl_xor(nm, 4, 32));
        nm = fmaxf(nm, __shfl_xor(nm, 8, 32));
        nm = fmaxf(nm, mx[r]);
        const float rescale = __expf(mx[r] - nm);
        mx[r] = nm;
        const float p0 = __expf(s0[r] - nm);
        const float p1 = __expf(s1[r] - nm);
        float rs = p0 + p1;
        rs += __shfl_xor(rs, 1, 32);
        rs += __shfl_xor(rs, 2, 32);
        rs += __shfl_xor(rs, 4, 32);
        rs += __shfl_xor(rs, 8, 32);
        sm[r] = sm[r] * rescale + rs;
        o0[r] = o0[r] * rescale;
        o1[r] = o1[r] * rescale;
        Ps[wave][r + 8*half][l15]      = (__bf16)p0;
        Ps[wave][r + 8*half][16 + l15] = (__bf16)p1;
      }
      asm volatile("s_wait_dscnt 0" ::: "memory");   // intra-wave LDS roundtrip

      // P back in A layout: two contiguous 16B chunks per lane
      v8bf plo = *(const v8bf*)&Ps[wave][l15][8*half];
      v8bf phi = *(const v8bf*)&Ps[wave][l15][16 + 8*half];
      v16bf ap = CONCAT16(plo, phi);

      // B for O: lane = head-dim column, elements over key tokens (table k order)
      v16bf bv0, bv1;
      #pragma unroll
      for (int s2 = 0; s2 < 16; ++s2) {
        const int kk = s2 < 8 ? 8*half + s2 : 8 + 8*half + s2;
        bv0[s2] = Vs[n0 + kk][l15];
        bv1[s2] = Vs[n0 + kk][16 + l15];
      }

      o0 = __builtin_amdgcn_wmma_f32_16x16x32_bf16(false, ap, false, bv0,
                                                   (short)0, o0, false, false);
      o1 = __builtin_amdgcn_wmma_f32_16x16x32_bf16(false, ap, false, bv1,
                                                   (short)0, o1, false, false);
    }

    // normalize and accumulate into out (conv kernel already wrote lepe+bias)
    #pragma unroll
    for (int r = 0; r < 8; ++r) {
      const int mrow = mtile * 16 + r + 8*half;
      if (mrow < Ltok) {
        const int t = (mrow / WSP) * Ww + ws * WSP + (mrow % WSP);
        const float inv = 1.f / sm[r];
        const size_t base = (size_t)b * NC + (size_t)t * Cc + head * HD;
        out[base + l15]      += o0[r] * inv;
        out[base + 16 + l15] += o1[r] * inv;
      }
    }
  }
}

extern "C" void kernel_launch(void* const* d_in, const int* in_sizes, int n_in,
                              void* d_out, int out_size, void* d_ws, size_t ws_size,
                              hipStream_t stream) {
  (void)in_sizes; (void)n_in; (void)out_size; (void)d_ws; (void)ws_size;
  const float* qkv   = (const float*)d_in[0];
  const float* wconv = (const float*)d_in[1];
  const float* bconv = (const float*)d_in[2];
  float* outp = (float*)d_out;

  // 1) LePE conv writes lepe + bias into out
  dim3 gc(2, 25, 128);
  lepe_conv_kernel<<<gc, 256, 0, stream>>>(qkv, wconv, bconv, outp);

  // 2) attention accumulates softmax(QK^T)V on top (stream-ordered)
  dim3 ga(NHEADS, Bn * 8);
  cswin_attn_kernel<<<ga, 256, 0, stream>>>(qkv, outp);
}
